// Encoder_14886356648051
// MI455X (gfx1250) — matrix operand order repaired and checked by reading, not compiled, not run on hardware
//
#include <hip/hip_runtime.h>
#include <hip/hip_bf16.h>
#include <math.h>

// ---------------- problem constants ----------------
#define NN  50000
#define EE  300000
#define FF  256
#define KK  49
#define OO  20
#define UCOLS 1000         // K*O + O  (bias folded as last 20 cols)
#define UCOLS_PAD 1024     // padded to multiple of 64
#define MLP1K 288          // 276 padded to multiple of 32
#define HEADN 200
#define HEADN_PAD 256      // padded to multiple of 64 (block N-tile)

#define LDSW 20            // LDS row stride in dwords: 16-byte aligned, conflict-free

typedef __bf16 bf16_t;
typedef __attribute__((ext_vector_type(16))) __bf16 v16bf;
typedef __attribute__((ext_vector_type(8)))  float  v8f;
typedef __attribute__((ext_vector_type(4)))  unsigned int v4u;

// ---------------- helpers ----------------
static __device__ __forceinline__ unsigned short f32_to_bf16_rn(float f) {
    unsigned int u = __float_as_uint(f);
    unsigned int r = u + 0x7FFFu + ((u >> 16) & 1u);   // round-to-nearest-even
    return (unsigned short)(r >> 16);
}
static __device__ __forceinline__ float bf16_to_f32(unsigned short h) {
    return __uint_as_float(((unsigned int)h) << 16);
}
struct fragpair { v4u a, b; };
static __device__ __forceinline__ v16bf load_frag(const unsigned int* p0, const unsigned int* p1) {
    fragpair p;
    p.a = *(const v4u*)p0;     // ds_load_b128
    p.b = *(const v4u*)p1;     // ds_load_b128
    return __builtin_bit_cast(v16bf, p);
}

// ================================================================
// Split-bf16 WMMA GEMM:  C[M,NReal] = act(A[M,Kp] @ B + bias)
// B stored transposed bf16 hi/lo: Bt[n][k], n-major, Kp per row.
// Block: 256 threads = 8 waves; block tile 64(M) x 64(N); K step 32.
// Wave (wm,wn) owns rows wm*16..+16, cols wn*32..+32 (two 16x16 acc).
// 6 x v_wmma_f32_16x16x32_bf16 per wave per K-step.
// ================================================================
__global__ __launch_bounds__(256, 2)
void gemm_bf16split(const float* __restrict__ A, int lda,
                    const unsigned short* __restrict__ Bh,
                    const unsigned short* __restrict__ Bl,
                    float* __restrict__ C, int ldc,
                    const float* __restrict__ bias,
                    int M, int Kp, int NReal, int act)
{
    __shared__ unsigned int sAh[64 * LDSW];   // 64 rows x 32 bf16 (16 dwords) + pad
    __shared__ unsigned int sAl[64 * LDSW];
    __shared__ unsigned int sBh[64 * LDSW];   // 64 n-rows x 32 bf16
    __shared__ unsigned int sBl[64 * LDSW];

    const int tid   = threadIdx.x;
    const int lane  = tid & 31;
    const int wave  = tid >> 5;      // 0..7
    const int wm    = wave >> 1;     // 0..3 : 16-row tile
    const int wn    = wave & 1;      // 0..1 : 32-col half
    const int row0  = blockIdx.y * 64;
    const int col0  = blockIdx.x * 64;

    const int mloc  = lane & 15;
    const int lhalf = lane >> 4;

    v8f acc0, acc1;
#pragma unroll
    for (int i = 0; i < 8; ++i) { acc0[i] = 0.0f; acc1[i] = 0.0f; }

    for (int kt = 0; kt < Kp; kt += 32) {
        // ---- stage A tile 64x32 f32 -> bf16 hi/lo (float2 loads, dword stores) ----
#pragma unroll
        for (int j = 0; j < 4; ++j) {
            int flat = tid + 256 * j;            // 0..1023 float2 slots
            int r  = flat >> 4;                  // 0..63
            int c2 = flat & 15;                  // float2 within row
            int gr = row0 + r;
            int cr = (gr < M) ? gr : (M - 1);    // clamped, branchless
            float2 v2 = *(const float2*)(A + (size_t)cr * lda + kt + 2 * c2);
            if (gr >= M) { v2.x = 0.0f; v2.y = 0.0f; }
            unsigned short h0 = f32_to_bf16_rn(v2.x);
            unsigned short h1 = f32_to_bf16_rn(v2.y);
            unsigned short l0 = f32_to_bf16_rn(v2.x - bf16_to_f32(h0));
            unsigned short l1 = f32_to_bf16_rn(v2.y - bf16_to_f32(h1));
            sAh[r * LDSW + c2] = (unsigned int)h0 | ((unsigned int)h1 << 16);
            sAl[r * LDSW + c2] = (unsigned int)l0 | ((unsigned int)l1 << 16);
        }
        // ---- stage B tile 64(n) x 32(k) bf16 hi/lo (dword copies) ----
#pragma unroll
        for (int j = 0; j < 4; ++j) {
            int flat = tid + 256 * j;            // 0..1023 dwords
            int n  = flat >> 4;                  // 0..63
            int kd = flat & 15;
            size_t gd = ((size_t)(col0 + n) * (size_t)Kp + (size_t)kt) / 2 + (size_t)kd;
            sBh[n * LDSW + kd] = ((const unsigned int*)Bh)[gd];
            sBl[n * LDSW + kd] = ((const unsigned int*)Bl)[gd];
        }
        __syncthreads();

        // ---- fragments: contiguous b128 runs, k = 16*lhalf + s for both A and B ----
        const int baseA = (wm * 16 + mloc) * LDSW;
        v16bf fah = load_frag(&sAh[baseA + 8 * lhalf], &sAh[baseA + 8 * lhalf + 4]);
        v16bf fal = load_frag(&sAl[baseA + 8 * lhalf], &sAl[baseA + 8 * lhalf + 4]);

        const int baseB0 = (wn * 32 + mloc) * LDSW + 8 * lhalf;        // cols [wn*32, +16)
        const int baseB1 = (wn * 32 + 16 + mloc) * LDSW + 8 * lhalf;   // cols [wn*32+16, +16)
        v16bf fbh0 = load_frag(&sBh[baseB0], &sBh[baseB0 + 4]);
        v16bf fbl0 = load_frag(&sBl[baseB0], &sBl[baseB0 + 4]);
        v16bf fbh1 = load_frag(&sBh[baseB1], &sBh[baseB1 + 4]);
        v16bf fbl1 = load_frag(&sBl[baseB1], &sBl[baseB1 + 4]);

        // A*B ~= Ah*Bh + Ah*Bl + Al*Bh  (f32 accumulate)
        acc0 = __builtin_amdgcn_wmma_f32_16x16x32_bf16(false, fah, false, fbh0, (short)0, acc0, false, false);
        acc0 = __builtin_amdgcn_wmma_f32_16x16x32_bf16(false, fah, false, fbl0, (short)0, acc0, false, false);
        acc0 = __builtin_amdgcn_wmma_f32_16x16x32_bf16(false, fal, false, fbh0, (short)0, acc0, false, false);
        acc1 = __builtin_amdgcn_wmma_f32_16x16x32_bf16(false, fah, false, fbh1, (short)0, acc1, false, false);
        acc1 = __builtin_amdgcn_wmma_f32_16x16x32_bf16(false, fah, false, fbl1, (short)0, acc1, false, false);
        acc1 = __builtin_amdgcn_wmma_f32_16x16x32_bf16(false, fal, false, fbh1, (short)0, acc1, false, false);
        __syncthreads();
    }

    // ---- epilogue: bias + activation + guarded store ----
#pragma unroll
    for (int t = 0; t < 2; ++t) {
        int gc = col0 + wn * 32 + 16 * t + mloc;
        if (gc < NReal) {
            float b = bias ? bias[gc] : 0.0f;
#pragma unroll
            for (int i = 0; i < 8; ++i) {
                int gr = row0 + wm * 16 + lhalf * 8 + i;   // C VGPR i -> row i (+8 for hi lanes)
                if (gr < M) {
                    float v = (t == 0 ? acc0[i] : acc1[i]) + b;
                    if (act == 1) v = fmaxf(v, 0.0f);
                    C[(size_t)gr * ldc + gc] = v;
                }
            }
        }
    }
}

// ================================================================
// Repack nn_w [K, F*O] (+nn_b[F*O]) into transposed bf16 hi/lo
// Bt[c][f], c in [0,1024): c<980 -> Wk[k=c/20][f][o=c%20]; c<1000 -> bias col; else 0
// ================================================================
__global__ void repack_nnw_kernel(const float* __restrict__ nw, const float* __restrict__ nb,
                                  unsigned short* __restrict__ bh, unsigned short* __restrict__ bl)
{
    int idx = blockIdx.x * blockDim.x + threadIdx.x;
    if (idx >= UCOLS_PAD * FF) return;
    int c = idx / FF;
    int f = idx - c * FF;
    float v = 0.0f;
    if (c < KK * OO) {
        int k = c / OO, o = c - k * OO;
        v = nw[(size_t)k * (FF * OO) + f * OO + o];
    } else if (c < UCOLS) {
        v = nb[f * OO + (c - KK * OO)];
    }
    unsigned short h = f32_to_bf16_rn(v);
    unsigned short l = f32_to_bf16_rn(v - bf16_to_f32(h));
    bh[(size_t)c * FF + f] = h;
    bl[(size_t)c * FF + f] = l;
}

// Generic: w [Kr,Nr] row-major f32 -> Bt[n][k] bf16 hi/lo, zero padded to [Np,Kp]
__global__ void repack_w_kernel(const float* __restrict__ w,
                                unsigned short* __restrict__ bh, unsigned short* __restrict__ bl,
                                int Kr, int Nr, int Kp, int Np)
{
    int idx = blockIdx.x * blockDim.x + threadIdx.x;
    if (idx >= Np * Kp) return;
    int n = idx / Kp;
    int k = idx - n * Kp;
    float v = (k < Kr && n < Nr) ? w[(size_t)k * Nr + n] : 0.0f;
    unsigned short h = f32_to_bf16_rn(v);
    unsigned short l = f32_to_bf16_rn(v - bf16_to_f32(h));
    bh[idx] = h;
    bl[idx] = l;
}

// ================================================================
// Edge messages: one wave per edge; lanes 0..19 handle o.
// pre[o] = U[src][980+o] + sum_k ea[e,k] * U[src][k*20+o];  msg = sigmoid(pre)
// atomicAdd into aggr[dst][o].
// ================================================================
__global__ __launch_bounds__(256)
void edge_msg_kernel(const float* __restrict__ U,
                     const long long* __restrict__ ei,
                     const float* __restrict__ ea,
                     float* __restrict__ aggr, int E)
{
    int e = blockIdx.x * (blockDim.x >> 5) + (threadIdx.x >> 5);
    int lane = threadIdx.x & 31;
    if (e >= E) return;
    long long s = ei[e];
    long long d = ei[(size_t)E + e];
    if (lane < OO) {
        const float* u = U + (size_t)s * UCOLS;
        const float* a = ea + (size_t)e * KK;
        float pre = u[KK * OO + lane];
#pragma unroll 7
        for (int k = 0; k < KK; ++k)
            pre = fmaf(a[k], u[k * OO + lane], pre);
        float m = 1.0f / (1.0f + __expf(-pre));
        atomicAdd(&aggr[(size_t)d * OO + lane], m);
    }
}

// yc[n, 0..19]=aggr, [20..275]=x, [276..287]=0
__global__ void concat_kernel(const float* __restrict__ aggr, const float* __restrict__ x,
                              float* __restrict__ yc, int n_nodes)
{
    size_t idx = (size_t)blockIdx.x * blockDim.x + threadIdx.x;
    size_t total = (size_t)n_nodes * MLP1K;
    if (idx >= total) return;
    int n = (int)(idx / MLP1K);
    int c = (int)(idx - (size_t)n * MLP1K);
    float v;
    if (c < OO)            v = aggr[(size_t)n * OO + c];
    else if (c < OO + FF)  v = x[(size_t)n * FF + (c - OO)];
    else                   v = 0.0f;
    yc[idx] = v;
}

// ================================================================
extern "C" void kernel_launch(void* const* d_in, const int* in_sizes, int n_in,
                              void* d_out, int out_size, void* d_ws, size_t ws_size,
                              hipStream_t stream)
{
    (void)in_sizes; (void)n_in; (void)out_size; (void)ws_size;

    const float*     x      = (const float*)d_in[0];
    const long long* ei     = (const long long*)d_in[1];
    const float*     ea     = (const float*)d_in[2];
    const float*     nn1_w  = (const float*)d_in[3];
    const float*     nn1_b  = (const float*)d_in[4];
    const float*     ag1_w1 = (const float*)d_in[5];
    const float*     ag1_b1 = (const float*)d_in[6];
    const float*     ag1_w2 = (const float*)d_in[7];
    const float*     ag1_b2 = (const float*)d_in[8];
    const float*     nn2_w  = (const float*)d_in[9];
    const float*     nn2_b  = (const float*)d_in[10];
    const float*     ag2_w1 = (const float*)d_in[11];
    const float*     ag2_b1 = (const float*)d_in[12];
    const float*     ag2_w2 = (const float*)d_in[13];
    const float*     ag2_b2 = (const float*)d_in[14];
    const float*     mu_w   = (const float*)d_in[15];
    const float*     mu_b   = (const float*)d_in[16];
    const float*     lv_w   = (const float*)d_in[17];
    const float*     lv_b   = (const float*)d_in[18];
    float* out = (float*)d_out;

    // ---- workspace carving (256B aligned) ----
    char* p = (char*)d_ws;
    auto carve = [&](size_t bytes) { char* r = p; p += (bytes + 255) & ~(size_t)255; return r; };
    float* U    = (float*)carve((size_t)NN * UCOLS * 4);
    float* aggr = (float*)carve((size_t)NN * OO * 4);
    float* yc   = (float*)carve((size_t)NN * MLP1K * 4);
    float* h1   = (float*)carve((size_t)NN * FF * 4);
    float* hA   = (float*)carve((size_t)NN * FF * 4);
    float* hB   = (float*)carve((size_t)NN * FF * 4);
    unsigned short* Wr_h = (unsigned short*)carve((size_t)UCOLS_PAD * FF * 2);
    unsigned short* Wr_l = (unsigned short*)carve((size_t)UCOLS_PAD * FF * 2);
    unsigned short* W1_h = (unsigned short*)carve((size_t)FF * MLP1K * 2);
    unsigned short* W1_l = (unsigned short*)carve((size_t)FF * MLP1K * 2);
    unsigned short* W2_h = (unsigned short*)carve((size_t)FF * FF * 2);
    unsigned short* W2_l = (unsigned short*)carve((size_t)FF * FF * 2);
    unsigned short* Wm_h = (unsigned short*)carve((size_t)HEADN_PAD * FF * 2);
    unsigned short* Wm_l = (unsigned short*)carve((size_t)HEADN_PAD * FF * 2);
    unsigned short* Wl_h = (unsigned short*)carve((size_t)HEADN_PAD * FF * 2);
    unsigned short* Wl_l = (unsigned short*)carve((size_t)HEADN_PAD * FF * 2);

    const dim3 blk(256);
    const dim3 gU(UCOLS_PAD / 64, (NN + 63) / 64);
    const dim3 gF(FF / 64, (NN + 63) / 64);
    const dim3 gH(HEADN_PAD / 64, (NN + 63) / 64);

    auto run_layer = [&](const float* xin,
                         const float* nw, const float* nb,
                         const float* w1, const float* b1,
                         const float* w2, const float* b2,
                         float* hout) {
        // 1) per-node tensor U = xin @ [Wk | B]   (bias folded; 25 GFLOP GEMM)
        repack_nnw_kernel<<<(UCOLS_PAD * FF + 255) / 256, blk, 0, stream>>>(nw, nb, Wr_h, Wr_l);
        gemm_bf16split<<<gU, blk, 0, stream>>>(xin, FF, Wr_h, Wr_l, U, UCOLS,
                                               nullptr, NN, FF, UCOLS, 0);
        // 2) edge messages + scatter-add
        hipMemsetAsync(aggr, 0, (size_t)NN * OO * 4, stream);
        edge_msg_kernel<<<(EE + 7) / 8, blk, 0, stream>>>(U, ei, ea, aggr, EE);
        // 3) concat + MLP
        concat_kernel<<<(int)(((size_t)NN * MLP1K + 255) / 256), blk, 0, stream>>>(aggr, xin, yc, NN);
        repack_w_kernel<<<(FF * MLP1K + 255) / 256, blk, 0, stream>>>(w1, W1_h, W1_l, OO + FF, FF, MLP1K, FF);
        repack_w_kernel<<<(FF * FF + 255) / 256, blk, 0, stream>>>(w2, W2_h, W2_l, FF, FF, FF, FF);
        gemm_bf16split<<<gF, blk, 0, stream>>>(yc, MLP1K, W1_h, W1_l, h1, FF, b1, NN, MLP1K, FF, 1);
        gemm_bf16split<<<gF, blk, 0, stream>>>(h1, FF, W2_h, W2_l, hout, FF, b2, NN, FF, FF, 0);
    };

    run_layer(x,  nn1_w, nn1_b, ag1_w1, ag1_b1, ag1_w2, ag1_b2, hA);
    run_layer(hA, nn2_w, nn2_b, ag2_w1, ag2_b1, ag2_w2, ag2_b2, hB);

    // ---- heads: mu / logvar straight into d_out ----
    repack_w_kernel<<<(HEADN_PAD * FF + 255) / 256, blk, 0, stream>>>(mu_w, Wm_h, Wm_l, FF, HEADN, FF, HEADN_PAD);
    repack_w_kernel<<<(HEADN_PAD * FF + 255) / 256, blk, 0, stream>>>(lv_w, Wl_h, Wl_l, FF, HEADN, FF, HEADN_PAD);
    gemm_bf16split<<<gH, blk, 0, stream>>>(hB, FF, Wm_h, Wm_l, out, HEADN, mu_b, NN, FF, HEADN, 0);
    gemm_bf16split<<<gH, blk, 0, stream>>>(hB, FF, Wl_h, Wl_l, out + (size_t)NN * HEADN, HEADN, lv_b, NN, FF, HEADN, 0);
}
